// AmorphousParticleGNN_27702539059817
// MI455X (gfx1250) — compile-verified, matrix operand order checked by reading, not run on hardware
//
#include <hip/hip_runtime.h>

#define N_PART 6000
#define HDIM   256
#define LLAYERS 10
#define KNN    30
#define PDIM   128
#define EDGES  (N_PART * KNN)

typedef __attribute__((ext_vector_type(16))) __bf16 v16bf;
typedef __attribute__((ext_vector_type(8)))  float  v8f;

__device__ __forceinline__ unsigned short f2bf(float f) {
    unsigned int u = __float_as_uint(f);
    u += 0x7FFFu + ((u >> 16) & 1u);          // round-to-nearest-even
    return (unsigned short)(u >> 16);
}

__device__ __forceinline__ v8f wmma_bf16(v16bf a, v16bf b, v8f c) {
    return __builtin_amdgcn_wmma_f32_16x16x32_bf16(false, a, false, b, (short)0, c, false, false);
}

// ---------------------------------------------------------------------------
// Pack f32 weight [Krows x Ncols] (row-major) into B-fragment layout:
// out[((kt*NT + nt)*32 + lane)*16 + j] = bf16(W[kt*32 + (lane>>4)*16 + j][nt*16 + (lane&15)])
// so each lane's 16x16x32 B-fragment is 16 contiguous bf16 (32 bytes).
// ---------------------------------------------------------------------------
__global__ void pack_kernel(const float* __restrict__ W, unsigned short* __restrict__ out,
                            int Krows, int Ncols) {
    int t = blockIdx.x * 256 + threadIdx.x;
    int total = Krows * Ncols;
    if (t >= total) return;
    int NT = Ncols >> 4;
    int j = t & 15;
    int rest = t >> 4;
    int lane = rest & 31; rest >>= 5;
    int nt = rest % NT;
    int kt = rest / NT;
    int k = kt * 32 + ((lane >> 4) << 4) + j;
    int n = nt * 16 + (lane & 15);
    out[t] = f2bf(W[(size_t)k * Ncols + n]);
}

// ---------------------------------------------------------------------------
// kNN with PBC minimum image. One thread per particle, top-30 insertion sort.
// ---------------------------------------------------------------------------
__global__ void knn_kernel(const float* __restrict__ pos, const float* __restrict__ box,
                           int* __restrict__ nbr, float* __restrict__ eattr,
                           float* __restrict__ counts) {
    int i = blockIdx.x * blockDim.x + threadIdx.x;
    if (i >= N_PART) return;
    float bx = box[0], by = box[1], bz = box[2];
    float px = pos[3 * i], py = pos[3 * i + 1], pz = pos[3 * i + 2];
    float bd[KNN]; int bi[KNN];
#pragma unroll
    for (int r = 0; r < KNN; r++) { bd[r] = 3.0e38f; bi[r] = 0; }
    for (int j = 0; j < N_PART; j++) {
        if (j == i) continue;
        float dx = px - pos[3 * j];     dx -= rintf(dx / bx) * bx;
        float dy = py - pos[3 * j + 1]; dy -= rintf(dy / by) * by;
        float dz = pz - pos[3 * j + 2]; dz -= rintf(dz / bz) * bz;
        float d2 = dx * dx + dy * dy + dz * dz;
        if (d2 < bd[KNN - 1]) {
            int p = KNN - 1;
            while (p > 0 && bd[p - 1] > d2) { bd[p] = bd[p - 1]; bi[p] = bi[p - 1]; p--; }
            bd[p] = d2; bi[p] = j;
        }
    }
    for (int r = 0; r < KNN; r++) {
        int j = bi[r];
        float dx = px - pos[3 * j];     dx -= rintf(dx / bx) * bx;
        float dy = py - pos[3 * j + 1]; dy -= rintf(dy / by) * by;
        float dz = pz - pos[3 * j + 2]; dz -= rintf(dz / bz) * bz;
        int e = i * KNN + r;
        nbr[e] = j;
        eattr[4 * e + 0] = dx;
        eattr[4 * e + 1] = dy;
        eattr[4 * e + 2] = dz;
        eattr[4 * e + 3] = sqrtf(bd[r]);
        atomicAdd(&counts[j], 1.0f);
    }
}

// ---------------------------------------------------------------------------
// Encoder: h = pos @ enc_W + enc_b    (K=3, pure VALU)
// ---------------------------------------------------------------------------
__global__ void enc_kernel(const float* __restrict__ pos, const float* __restrict__ W,
                           const float* __restrict__ b, float* __restrict__ h) {
    int idx = blockIdx.x * 256 + threadIdx.x;
    if (idx >= N_PART * HDIM) return;
    int n = idx / HDIM, c = idx % HDIM;
    h[idx] = b[c] + pos[3 * n] * W[c] + pos[3 * n + 1] * W[HDIM + c] + pos[3 * n + 2] * W[2 * HDIM + c];
}

// ---------------------------------------------------------------------------
// Node GEMM: C[6000,256] = A[6000,256] @ Wp (bf16 WMMA, f32 accumulate).
// Block = 8 waves, 16 rows x 256 cols; wave w -> column tiles 2w, 2w+1.
// ---------------------------------------------------------------------------
__global__ void __launch_bounds__(256)
gemm256_kernel(const float* __restrict__ A, const unsigned short* __restrict__ Wp,
               float* __restrict__ C) {
    __shared__ unsigned short As[16 * 256];
    int t = threadIdx.x;
    int row0 = blockIdx.x * 16;
    for (int idx = t; idx < 16 * 256; idx += 256)
        As[idx] = f2bf(A[(size_t)(row0 + (idx >> 8)) * 256 + (idx & 255)]);
    __syncthreads();
    int w = t >> 5, lane = t & 31;
    int arow = lane & 15, ak = (lane >> 4) << 4;
    int nt0 = 2 * w;
    v8f c0 = {}; v8f c1 = {};
#pragma unroll
    for (int ks = 0; ks < 8; ks++) {
        v16bf a  = *(const v16bf*)(As + arow * 256 + ks * 32 + ak);
        v16bf b0 = *(const v16bf*)(Wp + (((size_t)(ks * 16 + nt0) * 32 + lane) << 4));
        v16bf b1 = *(const v16bf*)(Wp + (((size_t)(ks * 16 + nt0 + 1) * 32 + lane) << 4));
        c0 = wmma_bf16(a, b0, c0);
        c1 = wmma_bf16(a, b1, c1);
    }
    int crow = 8 * (lane >> 4), col = lane & 15;
#pragma unroll
    for (int v = 0; v < 8; v++) {
        size_t r = (size_t)(row0 + crow + v) * 256;
        C[r + nt0 * 16 + col] = c0[v];
        C[r + (nt0 + 1) * 16 + col] = c1[v];
    }
}

// ---------------------------------------------------------------------------
// Edge pass: m = LN(relu(A1[dst] + A2[src] + e@W3 + b)); atomic agg[dst] += m.
// One wave per edge; LN stats via wave32 shuffles.
// ---------------------------------------------------------------------------
__global__ void __launch_bounds__(256)
edge_kernel(const float* __restrict__ A1, const float* __restrict__ A2,
            const float* __restrict__ eattr, const int* __restrict__ nbr,
            const float* __restrict__ mW, const float* __restrict__ mb,
            const float* __restrict__ mg, const float* __restrict__ mbt,
            float* __restrict__ agg) {
    int wave = threadIdx.x >> 5, lane = threadIdx.x & 31;
    int e = blockIdx.x * 8 + wave;
    if (e >= EDGES) return;
    int dst = nbr[e];
    int src = e / KNN;
    const float* r1 = A1 + (size_t)dst * HDIM;
    const float* r2 = A2 + (size_t)src * HDIM;
    float e0 = eattr[4 * e], e1 = eattr[4 * e + 1], e2 = eattr[4 * e + 2], e3 = eattr[4 * e + 3];
    const float* w3 = mW + 512 * HDIM;   // rows 512..515 of this layer's msg_W
    float val[8];
    float s = 0.f;
#pragma unroll
    for (int j = 0; j < 8; j++) {
        int col = lane + 32 * j;
        float v = r1[col] + r2[col] + mb[col]
                + e0 * w3[col] + e1 * w3[HDIM + col]
                + e2 * w3[2 * HDIM + col] + e3 * w3[3 * HDIM + col];
        v = fmaxf(v, 0.f);
        val[j] = v; s += v;
    }
#pragma unroll
    for (int off = 16; off > 0; off >>= 1) s += __shfl_xor(s, off, 32);
    float mu = s * (1.0f / HDIM);
    float q = 0.f;
#pragma unroll
    for (int j = 0; j < 8; j++) { float d = val[j] - mu; q += d * d; }
#pragma unroll
    for (int off = 16; off > 0; off >>= 1) q += __shfl_xor(q, off, 32);
    float rstd = rsqrtf(q * (1.0f / HDIM) + 1e-5f);
    float* ag = agg + (size_t)dst * HDIM;
#pragma unroll
    for (int j = 0; j < 8; j++) {
        int col = lane + 32 * j;
        atomicAdd(&ag[col], (val[j] - mu) * rstd * mg[col] + mbt[col]);
    }
}

// ---------------------------------------------------------------------------
// Update: u = LN(relu([h | agg/denom] @ uW + ub)); h += u.  K = 512.
// ---------------------------------------------------------------------------
__global__ void __launch_bounds__(256)
upd_kernel(float* __restrict__ h, const float* __restrict__ agg,
           const float* __restrict__ counts,
           const unsigned short* __restrict__ Wp,
           const float* __restrict__ ub, const float* __restrict__ ug,
           const float* __restrict__ ubt) {
    __shared__ unsigned short As[16 * 512];
    __shared__ float Cs[16 * 256];
    __shared__ float mu_s[16], rs_s[16];
    int t = threadIdx.x;
    int row0 = blockIdx.x * 16;
    for (int idx = t; idx < 16 * 512; idx += 256) {
        int r = idx >> 9, c = idx & 511;
        int row = row0 + r;
        float v;
        if (c < 256) v = h[(size_t)row * 256 + c];
        else         v = agg[(size_t)row * 256 + (c - 256)] / fmaxf(counts[row], 1.0f);
        As[idx] = f2bf(v);
    }
    __syncthreads();
    int w = t >> 5, lane = t & 31;
    int arow = lane & 15, ak = (lane >> 4) << 4;
    int nt0 = 2 * w;
    v8f c0 = {}; v8f c1 = {};
#pragma unroll
    for (int ks = 0; ks < 16; ks++) {
        v16bf a  = *(const v16bf*)(As + arow * 512 + ks * 32 + ak);
        v16bf b0 = *(const v16bf*)(Wp + (((size_t)(ks * 16 + nt0) * 32 + lane) << 4));
        v16bf b1 = *(const v16bf*)(Wp + (((size_t)(ks * 16 + nt0 + 1) * 32 + lane) << 4));
        c0 = wmma_bf16(a, b0, c0);
        c1 = wmma_bf16(a, b1, c1);
    }
    int crow = 8 * (lane >> 4), col = lane & 15;
#pragma unroll
    for (int v = 0; v < 8; v++) {
        int cc0 = nt0 * 16 + col, cc1 = cc0 + 16;
        Cs[(crow + v) * 256 + cc0] = fmaxf(c0[v] + ub[cc0], 0.f);
        Cs[(crow + v) * 256 + cc1] = fmaxf(c1[v] + ub[cc1], 0.f);
    }
    __syncthreads();
    if (t < 16) {
        float s = 0.f, q = 0.f;
        for (int c = 0; c < 256; c++) s += Cs[t * 256 + c];
        float m = s * (1.0f / 256.0f);
        for (int c = 0; c < 256; c++) { float d = Cs[t * 256 + c] - m; q += d * d; }
        mu_s[t] = m;
        rs_s[t] = rsqrtf(q * (1.0f / 256.0f) + 1e-5f);
    }
    __syncthreads();
    for (int idx = t; idx < 16 * 256; idx += 256) {
        int r = idx >> 8, c = idx & 255;
        h[(size_t)(row0 + r) * 256 + c] += (Cs[idx] - mu_s[r]) * rs_s[r] * ug[c] + ubt[c];
    }
}

// ---------------------------------------------------------------------------
// Fused projection: out = relu(h@W1 + b1) @ W2 + b2 (intermediate bf16 in LDS).
// ---------------------------------------------------------------------------
__global__ void __launch_bounds__(256)
proj_kernel(const float* __restrict__ h,
            const unsigned short* __restrict__ W1p, const float* __restrict__ b1,
            const unsigned short* __restrict__ W2p, const float* __restrict__ b2,
            float* __restrict__ out) {
    __shared__ unsigned short As[16 * 256];
    __shared__ unsigned short Ts[16 * 256];
    int t = threadIdx.x;
    int row0 = blockIdx.x * 16;
    for (int idx = t; idx < 16 * 256; idx += 256)
        As[idx] = f2bf(h[(size_t)(row0 + (idx >> 8)) * 256 + (idx & 255)]);
    __syncthreads();
    int w = t >> 5, lane = t & 31;
    int arow = lane & 15, ak = (lane >> 4) << 4;
    {   // phase 1: 256 cols
        v8f c0 = {}; v8f c1 = {};
        int nt0 = 2 * w;
#pragma unroll
        for (int ks = 0; ks < 8; ks++) {
            v16bf a  = *(const v16bf*)(As + arow * 256 + ks * 32 + ak);
            v16bf b0 = *(const v16bf*)(W1p + (((size_t)(ks * 16 + nt0) * 32 + lane) << 4));
            v16bf b1 = *(const v16bf*)(W1p + (((size_t)(ks * 16 + nt0 + 1) * 32 + lane) << 4));
            c0 = wmma_bf16(a, b0, c0);
            c1 = wmma_bf16(a, b1, c1);
        }
        int crow = 8 * (lane >> 4), col = lane & 15;
#pragma unroll
        for (int v = 0; v < 8; v++) {
            int cc0 = nt0 * 16 + col, cc1 = cc0 + 16;
            Ts[(crow + v) * 256 + cc0] = f2bf(fmaxf(c0[v] + b1[cc0], 0.f));
            Ts[(crow + v) * 256 + cc1] = f2bf(fmaxf(c1[v] + b1[cc1], 0.f));
        }
    }
    __syncthreads();
    {   // phase 2: 128 cols, 8 waves x 1 tile
        v8f c0 = {};
        int nt = w;
#pragma unroll
        for (int ks = 0; ks < 8; ks++) {
            v16bf a  = *(const v16bf*)(Ts + arow * 256 + ks * 32 + ak);
            v16bf b0 = *(const v16bf*)(W2p + (((size_t)(ks * 8 + nt) * 32 + lane) << 4));
            c0 = wmma_bf16(a, b0, c0);
        }
        int crow = 8 * (lane >> 4), col = lane & 15;
#pragma unroll
        for (int v = 0; v < 8; v++)
            out[(size_t)(row0 + crow + v) * PDIM + nt * 16 + col] = c0[v] + b2[nt * 16 + col];
    }
}

// ---------------------------------------------------------------------------
extern "C" void kernel_launch(void* const* d_in, const int* in_sizes, int n_in,
                              void* d_out, int out_size, void* d_ws, size_t ws_size,
                              hipStream_t stream) {
    const float* pos    = (const float*)d_in[0];
    const float* box    = (const float*)d_in[1];
    const float* enc_W  = (const float*)d_in[2];
    const float* enc_b  = (const float*)d_in[3];
    const float* msg_W  = (const float*)d_in[4];
    const float* msg_b  = (const float*)d_in[5];
    const float* msg_g  = (const float*)d_in[6];
    const float* msg_bt = (const float*)d_in[7];
    const float* upd_W  = (const float*)d_in[8];
    const float* upd_b  = (const float*)d_in[9];
    const float* upd_g  = (const float*)d_in[10];
    const float* upd_bt = (const float*)d_in[11];
    const float* pW1    = (const float*)d_in[12];
    const float* pb1    = (const float*)d_in[13];
    const float* pW2    = (const float*)d_in[14];
    const float* pb2    = (const float*)d_in[15];
    float* out = (float*)d_out;

    char* ws = (char*)d_ws;
    size_t off = 0;
    auto alloc = [&](size_t bytes) -> char* {
        char* p = ws + off;
        off += (bytes + 255) & ~(size_t)255;
        return p;
    };
    float* h      = (float*)alloc(sizeof(float) * N_PART * HDIM);
    float* A1     = (float*)alloc(sizeof(float) * N_PART * HDIM);
    float* A2     = (float*)alloc(sizeof(float) * N_PART * HDIM);
    float* agg    = (float*)alloc(sizeof(float) * N_PART * HDIM);
    float* eattr  = (float*)alloc(sizeof(float) * EDGES * 4);
    int*   nbr    = (int*)  alloc(sizeof(int) * EDGES);
    float* counts = (float*)alloc(sizeof(float) * N_PART);
    unsigned short* Wm  = (unsigned short*)alloc(sizeof(unsigned short) * LLAYERS * 2 * 256 * 256);
    unsigned short* Wu  = (unsigned short*)alloc(sizeof(unsigned short) * LLAYERS * 512 * 256);
    unsigned short* Wp1 = (unsigned short*)alloc(sizeof(unsigned short) * 256 * 256);
    unsigned short* Wp2 = (unsigned short*)alloc(sizeof(unsigned short) * 256 * 128);

    hipMemsetAsync(counts, 0, sizeof(float) * N_PART, stream);

    for (int l = 0; l < LLAYERS; l++) {
        const float* mw = msg_W + (size_t)l * 516 * 256;
        pack_kernel<<<(256 * 256) / 256, 256, 0, stream>>>(mw,             Wm + (size_t)l * 131072,         256, 256);
        pack_kernel<<<(256 * 256) / 256, 256, 0, stream>>>(mw + 256 * 256, Wm + (size_t)l * 131072 + 65536, 256, 256);
        pack_kernel<<<(512 * 256) / 256, 256, 0, stream>>>(upd_W + (size_t)l * 512 * 256, Wu + (size_t)l * 131072, 512, 256);
    }
    pack_kernel<<<(256 * 256) / 256, 256, 0, stream>>>(pW1, Wp1, 256, 256);
    pack_kernel<<<(256 * 128) / 256, 256, 0, stream>>>(pW2, Wp2, 256, 128);

    knn_kernel<<<(N_PART + 255) / 256, 256, 0, stream>>>(pos, box, nbr, eattr, counts);
    enc_kernel<<<(N_PART * HDIM) / 256, 256, 0, stream>>>(pos, enc_W, enc_b, h);

    for (int l = 0; l < LLAYERS; l++) {
        hipMemsetAsync(agg, 0, sizeof(float) * N_PART * HDIM, stream);
        gemm256_kernel<<<N_PART / 16, 256, 0, stream>>>(h, Wm + (size_t)l * 131072,         A1);
        gemm256_kernel<<<N_PART / 16, 256, 0, stream>>>(h, Wm + (size_t)l * 131072 + 65536, A2);
        edge_kernel<<<EDGES / 8, 256, 0, stream>>>(A1, A2, eattr, nbr,
            msg_W + (size_t)l * 516 * 256, msg_b + l * HDIM, msg_g + l * HDIM, msg_bt + l * HDIM, agg);
        upd_kernel<<<N_PART / 16, 256, 0, stream>>>(h, agg, counts,
            Wu + (size_t)l * 131072, upd_b + l * HDIM, upd_g + l * HDIM, upd_bt + l * HDIM);
    }
    proj_kernel<<<N_PART / 16, 256, 0, stream>>>(h, Wp1, pb1, Wp2, pb2, out);
}
